// Seq2Seq_48086453846357
// MI455X (gfx1250) — matrix operand order repaired and checked by reading, not compiled, not run on hardware
//
#include <hip/hip_runtime.h>
#include <hip/hip_bf16.h>
#include <cstdint>

#define DEV __device__ __forceinline__

typedef __attribute__((ext_vector_type(16))) __bf16 v16bf;
typedef __attribute__((ext_vector_type(8)))  float  v8f;
typedef unsigned short u16t;
typedef unsigned int   u32t;

// Problem sizes (fixed by reference)
constexpr int Vv = 32000, Ee = 512, Hh = 1024, Bb = 32, Ss = 128, Tt = 128;
constexpr int G3H = 3 * Hh;
constexpr long long TV = (long long)Tt * Vv;
constexpr int NBLK_FC = Vv / 256;  // 125 blocks over the vocab

DEV u16t f2bf(float f) {
  u32t x = __builtin_bit_cast(u32t, f);
  u32t r = x + 0x7FFFu + ((x >> 16) & 1u);   // round-to-nearest-even
  return (u16t)(r >> 16);
}

DEV v8f zero8() { v8f z = {0.f,0.f,0.f,0.f,0.f,0.f,0.f,0.f}; return z; }

union PackU { u32t u[8]; uint4 q[2]; v16bf v; };

// A-tile (16x32 bf16) in contiguous k order (same as packB), from row-major bf16 source.
// lane L (L%16 = row): k = e + (L/16)*16 for element e.
DEV v16bf packA_bf16(const u16t* A, int ldk, int row, int kb, int lane) {
  int half = lane >> 4;
  const u16t* base = A + (size_t)row * ldk + kb + half * 16;
  PackU U;
#pragma unroll
  for (int p = 0; p < 8; ++p) {
    int off = 2 * p;
    U.u[p] = *(const u32t*)(base + off);
  }
  return U.v;
}

// B-tile (32x16 bf16): B^T(k,n) = W(n,k), W is (N x K) row-major bf16.
// lane L (L%16 = col): k = e + (L/16)*16  -> 16 consecutive k = two b128 loads.
DEV v16bf packB(const u16t* W, int ldk, int ncol, int kb, int lane) {
  int half = lane >> 4;
  const u16t* base = W + (size_t)ncol * ldk + kb + half * 16;
  PackU U;
  U.q[0] = *(const uint4*)(base);
  U.q[1] = *(const uint4*)(base + 8);
  return U.v;
}

DEV v8f wmma_bf16(v16bf a, v16bf b, v8f c) {
  return __builtin_amdgcn_wmma_f32_16x16x32_bf16(false, a, false, b, (short)0, c,
                                                 false, false);
}

// ---------------- utility kernels ----------------

__global__ void cvt_f32_bf16_kernel(const float* __restrict__ src,
                                    u16t* __restrict__ dst, int n) {
  int i = blockIdx.x * blockDim.x + threadIdx.x;
  int stride = gridDim.x * blockDim.x;
  for (; i < n; i += stride) dst[i] = f2bf(src[i]);
}

__global__ void init_kernel(float* out, float* h0, float* h1,
                            u16t* h0bf, u16t* h1bf, int* tok, const int* trg) {
  int i = blockIdx.x * blockDim.x + threadIdx.x;
  int stride = gridDim.x * blockDim.x;
  for (int idx = i; idx < Bb * Vv; idx += stride) {      // zero t=0 logits slice
    int b = idx / Vv, v = idx - b * Vv;
    out[(size_t)b * TV + v] = 0.f;
  }
  for (int idx = i; idx < Bb * Hh; idx += stride) {
    h0[idx] = 0.f; h1[idx] = 0.f; h0bf[idx] = 0; h1bf[idx] = 0;
  }
  if (i < Bb) tok[i] = trg[i * Tt];                      // tok0 = trg[:,0]
}

// xs[(s*B+b)*E + e] = bf16(enc_emb[src[b,s], e])
__global__ void embed_src_kernel(const int* __restrict__ src,
                                 const float* __restrict__ emb,
                                 u16t* __restrict__ xs) {
  int i = blockIdx.x * blockDim.x + threadIdx.x;
  int stride = gridDim.x * blockDim.x;
  int total = Ss * Bb * Ee;
  for (; i < total; i += stride) {
    int e  = i & (Ee - 1);
    int sb = i >> 9;             // /E
    int b  = sb & (Bb - 1);
    int s  = sb >> 5;            // /B
    int t  = src[b * Ss + s];
    xs[i] = f2bf(emb[(size_t)t * Ee + e]);
  }
}

__global__ void embed_tok_kernel(const int* __restrict__ tok,
                                 const float* __restrict__ emb,
                                 u16t* __restrict__ x) {
  for (int idx = threadIdx.x; idx < Bb * Ee; idx += blockDim.x) {
    int b = idx >> 9, e = idx & (Ee - 1);
    x[idx] = f2bf(emb[(size_t)tok[b] * Ee + e]);
  }
}

// ---------------- big GEMM: gi_all = A(M,K) @ W(3H,K)^T + bias ----------------
// grid (12, M/32), block 256 (8 waves). Wave: 32 rows x 32 cols (2x2 tiles).
__global__ __launch_bounds__(256) void gemm_bias_kernel(
    const u16t* __restrict__ A, int K, const u16t* __restrict__ W,
    const float* __restrict__ bias, float* __restrict__ C) {
  int lane = threadIdx.x & 31;
  int wave = threadIdx.x >> 5;
  int l15  = lane & 15;
  int nbase = blockIdx.x * 256 + wave * 32;
  int mbase = blockIdx.y * 32;
  int n0 = nbase + l15, n1 = n0 + 16;
  v8f a00 = zero8(), a01 = zero8(), a10 = zero8(), a11 = zero8();
  for (int kb = 0; kb < K; kb += 32) {
    if (kb + 32 < K) {  // stream next weight block (global_prefetch_b8)
      __builtin_prefetch((const void*)(W + (size_t)n0 * K + kb + 32), 0, 1);
      __builtin_prefetch((const void*)(W + (size_t)n1 * K + kb + 32), 0, 1);
    }
    v16bf A0 = packA_bf16(A, K, mbase + l15, kb, lane);
    v16bf A1 = packA_bf16(A, K, mbase + 16 + l15, kb, lane);
    v16bf B0 = packB(W, K, n0, kb, lane);
    v16bf B1 = packB(W, K, n1, kb, lane);
    a00 = wmma_bf16(A0, B0, a00);
    a10 = wmma_bf16(A1, B0, a10);
    a01 = wmma_bf16(A0, B1, a01);
    a11 = wmma_bf16(A1, B1, a11);
  }
  float b0 = bias[n0], b1 = bias[n1];
#pragma unroll
  for (int i = 0; i < 8; ++i) {
    int r0 = mbase + ((lane < 16) ? i : i + 8);
    int r1 = r0 + 16;
    C[(size_t)r0 * G3H + n0] = a00[i] + b0;
    C[(size_t)r0 * G3H + n1] = a01[i] + b1;
    C[(size_t)r1 * G3H + n0] = a10[i] + b0;
    C[(size_t)r1 * G3H + n1] = a11[i] + b1;
  }
}

// ---------------- GRU step ----------------
// XMODE 0: gi precomputed (incl. bih).  1: fuse gi = x @ Wih^T (x bf16).
// grid 8 blocks x 256 threads; wave handles 16 h-columns, all 32 batch rows.
// hbf is the bf16 copy of h (A operand); h stays f32 for the elementwise update.
// Writes h_new (f32) and hnew_bf (bf16 copy for the next consumer).
template <int XMODE>
__global__ __launch_bounds__(256) void gru_step_kernel(
    const u16t* __restrict__ xp, int Kx, const u16t* __restrict__ Wih,
    const float* __restrict__ bih, const float* __restrict__ gi,
    const float* __restrict__ h, const u16t* __restrict__ hbf,
    const u16t* __restrict__ Whh, const float* __restrict__ bhh,
    float* __restrict__ h_new, u16t* __restrict__ hnew_bf) {
  int lane = threadIdx.x & 31;
  int wave = threadIdx.x >> 5;
  int l15  = lane & 15;
  int jbase = blockIdx.x * 128 + wave * 16;
  int colR = jbase + l15;
  int colZ = Hh + colR;
  int colN = 2 * Hh + colR;

  v8f aR0 = zero8(), aR1 = zero8(), aZ0 = zero8(), aZ1 = zero8();
  v8f hN0 = zero8(), hN1 = zero8(), iN0 = zero8(), iN1 = zero8();

  // gh = h @ Whh^T  (K = H)
  for (int kb = 0; kb < Hh; kb += 32) {
    v16bf A0 = packA_bf16(hbf, Hh, l15, kb, lane);
    v16bf A1 = packA_bf16(hbf, Hh, 16 + l15, kb, lane);
    v16bf Br = packB(Whh, Hh, colR, kb, lane);
    v16bf Bz = packB(Whh, Hh, colZ, kb, lane);
    v16bf Bn = packB(Whh, Hh, colN, kb, lane);
    aR0 = wmma_bf16(A0, Br, aR0); aR1 = wmma_bf16(A1, Br, aR1);
    aZ0 = wmma_bf16(A0, Bz, aZ0); aZ1 = wmma_bf16(A1, Bz, aZ1);
    hN0 = wmma_bf16(A0, Bn, hN0); hN1 = wmma_bf16(A1, Bn, hN1);
  }
  if (XMODE != 0) {  // fused gi = x @ Wih^T (r,z sum into same acc; n kept apart)
    for (int kb = 0; kb < Kx; kb += 32) {
      v16bf A0 = packA_bf16(xp, Kx, l15, kb, lane);
      v16bf A1 = packA_bf16(xp, Kx, 16 + l15, kb, lane);
      v16bf Br = packB(Wih, Kx, colR, kb, lane);
      v16bf Bz = packB(Wih, Kx, colZ, kb, lane);
      v16bf Bn = packB(Wih, Kx, colN, kb, lane);
      aR0 = wmma_bf16(A0, Br, aR0); aR1 = wmma_bf16(A1, Br, aR1);
      aZ0 = wmma_bf16(A0, Bz, aZ0); aZ1 = wmma_bf16(A1, Bz, aZ1);
      iN0 = wmma_bf16(A0, Bn, iN0); iN1 = wmma_bf16(A1, Bn, iN1);
    }
  }
  float bhR = bhh[colR], bhZ = bhh[colZ], bhN = bhh[colN];
  float biR = 0.f, biZ = 0.f, biN = 0.f;
  if (XMODE != 0) { biR = bih[colR]; biZ = bih[colZ]; biN = bih[colN]; }

#pragma unroll
  for (int mt = 0; mt < 2; ++mt) {
#pragma unroll
    for (int i = 0; i < 8; ++i) {
      int m = mt * 16 + ((lane < 16) ? i : i + 8);
      float gr  = (mt ? aR1[i] : aR0[i]) + bhR;
      float gz  = (mt ? aZ1[i] : aZ0[i]) + bhZ;
      float ghn = (mt ? hN1[i] : hN0[i]) + bhN;
      float gin;
      if (XMODE != 0) {
        gr += biR; gz += biZ;
        gin = (mt ? iN1[i] : iN0[i]) + biN;
      } else {
        const float* grow = gi + (size_t)m * G3H;
        gr += grow[colR]; gz += grow[colZ]; gin = grow[colN];
      }
      float hv = h[(size_t)m * Hh + colR];
      float r = 1.f / (1.f + expf(-gr));
      float z = 1.f / (1.f + expf(-gz));
      float n = tanhf(gin + r * ghn);
      float o = (1.f - z) * n + z * hv;
      h_new[(size_t)m * Hh + colR]   = o;
      hnew_bf[(size_t)m * Hh + colR] = f2bf(o);
    }
  }
}

// ---------------- FC (logits) + per-block argmax ----------------
// grid 125 blocks x 256 threads; block covers 256 vocab columns.
__global__ __launch_bounds__(256) void fc_argmax_kernel(
    const u16t* __restrict__ h2bf, const u16t* __restrict__ W,
    const float* __restrict__ fcb, float* __restrict__ out, int t,
    float* __restrict__ pvals, int* __restrict__ pidx) {
  __shared__ float sv[32][8];
  __shared__ int   si[32][8];
  int lane = threadIdx.x & 31;
  int wave = threadIdx.x >> 5;
  int l15  = lane & 15;
  int nbase = blockIdx.x * 256 + wave * 32;
  int n0 = nbase + l15, n1 = n0 + 16;
  v8f a00 = zero8(), a01 = zero8(), a10 = zero8(), a11 = zero8();
  for (int kb = 0; kb < Hh; kb += 32) {
    if (kb + 32 < Hh) {
      __builtin_prefetch((const void*)(W + (size_t)n0 * Hh + kb + 32), 0, 1);
      __builtin_prefetch((const void*)(W + (size_t)n1 * Hh + kb + 32), 0, 1);
    }
    v16bf A0 = packA_bf16(h2bf, Hh, l15, kb, lane);
    v16bf A1 = packA_bf16(h2bf, Hh, 16 + l15, kb, lane);
    v16bf B0 = packB(W, Hh, n0, kb, lane);
    v16bf B1 = packB(W, Hh, n1, kb, lane);
    a00 = wmma_bf16(A0, B0, a00);
    a10 = wmma_bf16(A1, B0, a10);
    a01 = wmma_bf16(A0, B1, a01);
    a11 = wmma_bf16(A1, B1, a11);
  }
  float b0 = fcb[n0], b1 = fcb[n1];
#pragma unroll
  for (int mt = 0; mt < 2; ++mt) {
#pragma unroll
    for (int i = 0; i < 8; ++i) {
      int row = mt * 16 + ((lane < 16) ? i : i + 8);
      float v0 = (mt ? a10[i] : a00[i]) + b0;
      float v1 = (mt ? a11[i] : a01[i]) + b1;
      float* orow = out + (size_t)row * TV + (size_t)t * Vv;
      orow[n0] = v0;
      orow[n1] = v1;
      float bv = v0; int bi = n0;
      if (v1 > bv) { bv = v1; bi = n1; }
#pragma unroll
      for (int mask = 1; mask <= 8; mask <<= 1) {  // reduce 16-lane row group
        float ov = __shfl_xor(bv, mask, 32);
        int   oi = __shfl_xor(bi, mask, 32);
        if (ov > bv || (ov == bv && oi < bi)) { bv = ov; bi = oi; }
      }
      if (l15 == 0) { sv[row][wave] = bv; si[row][wave] = bi; }
    }
  }
  __syncthreads();
  if (threadIdx.x < 32) {
    int row = threadIdx.x;
    float bv = sv[row][0]; int bi = si[row][0];
#pragma unroll
    for (int w = 1; w < 8; ++w) {
      float ov = sv[row][w]; int oi = si[row][w];
      if (ov > bv || (ov == bv && oi < bi)) { bv = ov; bi = oi; }
    }
    pvals[row * NBLK_FC + blockIdx.x] = bv;
    pidx [row * NBLK_FC + blockIdx.x] = bi;
  }
}

__global__ void argmax_reduce_kernel(const float* __restrict__ pvals,
                                     const int* __restrict__ pidx,
                                     int* __restrict__ tok) {
  int row = threadIdx.x;
  if (row < Bb) {
    float bv = pvals[row * NBLK_FC]; int bi = pidx[row * NBLK_FC];
    for (int p = 1; p < NBLK_FC; ++p) {
      float ov = pvals[row * NBLK_FC + p]; int oi = pidx[row * NBLK_FC + p];
      if (ov > bv || (ov == bv && oi < bi)) { bv = ov; bi = oi; }
    }
    tok[row] = bi;
  }
}

// ---------------- host ----------------

extern "C" void kernel_launch(void* const* d_in, const int* in_sizes, int n_in,
                              void* d_out, int out_size, void* d_ws, size_t ws_size,
                              hipStream_t stream) {
  (void)in_sizes; (void)n_in; (void)out_size; (void)ws_size;
  const int*   src      = (const int*)  d_in[0];
  const int*   trg      = (const int*)  d_in[1];
  const float* enc_emb  = (const float*)d_in[3];
  const float* enc_Wih0 = (const float*)d_in[4];
  const float* enc_Whh0 = (const float*)d_in[5];
  const float* enc_bih0 = (const float*)d_in[6];
  const float* enc_bhh0 = (const float*)d_in[7];
  const float* enc_Wih1 = (const float*)d_in[8];
  const float* enc_Whh1 = (const float*)d_in[9];
  const float* enc_bih1 = (const float*)d_in[10];
  const float* enc_bhh1 = (const float*)d_in[11];
  const float* dec_emb  = (const float*)d_in[12];
  const float* dec_Wih0 = (const float*)d_in[13];
  const float* dec_Whh0 = (const float*)d_in[14];
  const float* dec_bih0 = (const float*)d_in[15];
  const float* dec_bhh0 = (const float*)d_in[16];
  const float* dec_Wih1 = (const float*)d_in[17];
  const float* dec_Whh1 = (const float*)d_in[18];
  const float* dec_bih1 = (const float*)d_in[19];
  const float* dec_bhh1 = (const float*)d_in[20];
  const float* fc_W     = (const float*)d_in[21];
  const float* fc_b     = (const float*)d_in[22];
  float* out = (float*)d_out;

  // workspace carve-up (256B aligned)
  char* p = (char*)d_ws;
  auto alloc = [&](size_t bytes) -> char* {
    char* r = p; p += (bytes + 255) & ~(size_t)255; return r;
  };
  u16t* wEih0 = (u16t*)alloc((size_t)G3H * Ee * 2);
  u16t* wEhh0 = (u16t*)alloc((size_t)G3H * Hh * 2);
  u16t* wEih1 = (u16t*)alloc((size_t)G3H * Hh * 2);
  u16t* wEhh1 = (u16t*)alloc((size_t)G3H * Hh * 2);
  u16t* wDih0 = (u16t*)alloc((size_t)G3H * Ee * 2);
  u16t* wDhh0 = (u16t*)alloc((size_t)G3H * Hh * 2);
  u16t* wDih1 = (u16t*)alloc((size_t)G3H * Hh * 2);
  u16t* wDhh1 = (u16t*)alloc((size_t)G3H * Hh * 2);
  u16t* wFc   = (u16t*)alloc((size_t)Vv * Hh * 2);
  u16t* xs    = (u16t*)alloc((size_t)Ss * Bb * Ee * 2);
  u16t* ys0   = (u16t*)alloc((size_t)Ss * Bb * Hh * 2);
  float* giA  = (float*)alloc((size_t)Ss * Bb * G3H * 4);
  float* h0A  = (float*)alloc((size_t)Bb * Hh * 4);
  float* h0B  = (float*)alloc((size_t)Bb * Hh * 4);
  float* h1A  = (float*)alloc((size_t)Bb * Hh * 4);
  float* h1B  = (float*)alloc((size_t)Bb * Hh * 4);
  u16t* h0Abf = (u16t*)alloc((size_t)Bb * Hh * 2);
  u16t* h0Bbf = (u16t*)alloc((size_t)Bb * Hh * 2);
  u16t* h1Abf = (u16t*)alloc((size_t)Bb * Hh * 2);
  u16t* h1Bbf = (u16t*)alloc((size_t)Bb * Hh * 2);
  u16t*  xdec = (u16t*) alloc((size_t)Bb * Ee * 2);
  float* pvals= (float*)alloc((size_t)Bb * NBLK_FC * 4);
  int*   pidxs= (int*)  alloc((size_t)Bb * NBLK_FC * 4);
  int*   tok  = (int*)  alloc((size_t)Bb * 4);

  // 1) weights -> bf16 (all fit in 192MB L2 afterwards)
  cvt_f32_bf16_kernel<<<512, 256, 0, stream>>>(enc_Wih0, wEih0, G3H * Ee);
  cvt_f32_bf16_kernel<<<512, 256, 0, stream>>>(enc_Whh0, wEhh0, G3H * Hh);
  cvt_f32_bf16_kernel<<<512, 256, 0, stream>>>(enc_Wih1, wEih1, G3H * Hh);
  cvt_f32_bf16_kernel<<<512, 256, 0, stream>>>(enc_Whh1, wEhh1, G3H * Hh);
  cvt_f32_bf16_kernel<<<512, 256, 0, stream>>>(dec_Wih0, wDih0, G3H * Ee);
  cvt_f32_bf16_kernel<<<512, 256, 0, stream>>>(dec_Whh0, wDhh0, G3H * Hh);
  cvt_f32_bf16_kernel<<<512, 256, 0, stream>>>(dec_Wih1, wDih1, G3H * Hh);
  cvt_f32_bf16_kernel<<<512, 256, 0, stream>>>(dec_Whh1, wDhh1, G3H * Hh);
  cvt_f32_bf16_kernel<<<1024, 256, 0, stream>>>(fc_W, wFc, Vv * Hh);

  init_kernel<<<2048, 256, 0, stream>>>(out, h0A, h1A, h0Abf, h1Abf, tok, trg);
  embed_src_kernel<<<2048, 256, 0, stream>>>(src, enc_emb, xs);

  // 2) encoder layer 0: hoisted gi GEMM (M=4096), then 128-step scan.
  //    ys0 slices double as the bf16 h-copy chain.
  gemm_bias_kernel<<<dim3(12, Ss * Bb / 32), 256, 0, stream>>>(xs, Ee, wEih0,
                                                               enc_bih0, giA);
  float* hc = h0A; float* hn = h0B;
  const u16t* hbf0c = h0Abf;            // zeros
  for (int s = 0; s < Ss; ++s) {
    u16t* hbf0n = ys0 + (size_t)s * Bb * Hh;
    gru_step_kernel<0><<<8, 256, 0, stream>>>(
        nullptr, 0, nullptr, nullptr, giA + (size_t)s * Bb * G3H, hc, hbf0c,
        wEhh0, enc_bhh0, hn, hbf0n);
    hbf0c = hbf0n;
    float* t2 = hc; hc = hn; hn = t2;
  }
  // 3) encoder layer 1
  gemm_bias_kernel<<<dim3(12, Ss * Bb / 32), 256, 0, stream>>>(ys0, Hh, wEih1,
                                                               enc_bih1, giA);
  float* hc1 = h1A; float* hn1 = h1B;
  u16t* hbf1c = h1Abf; u16t* hbf1n = h1Bbf;
  for (int s = 0; s < Ss; ++s) {
    gru_step_kernel<0><<<8, 256, 0, stream>>>(
        nullptr, 0, nullptr, nullptr, giA + (size_t)s * Bb * G3H, hc1, hbf1c,
        wEhh1, enc_bhh1, hn1, hbf1n);
    float* t2 = hc1; hc1 = hn1; hn1 = t2;
    u16t* t3 = hbf1c; hbf1c = hbf1n; hbf1n = t3;
  }

  // 4) greedy decoder: 127 steps, 5 launches/step (3 on the last).
  //    Layer-0 bf16 h continues from the last ys0 slice, then ping-pongs.
  u16t* hbf0n = h0Bbf;
  u16t* hbf0other = h0Abf;
  embed_tok_kernel<<<1, 256, 0, stream>>>(tok, dec_emb, xdec);
  for (int i = 0; i < Tt - 1; ++i) {
    gru_step_kernel<1><<<8, 256, 0, stream>>>(xdec, Ee, wDih0, dec_bih0, nullptr,
                                              hc, hbf0c, wDhh0, dec_bhh0, hn,
                                              hbf0n);
    gru_step_kernel<1><<<8, 256, 0, stream>>>(hbf0n, Hh, wDih1, dec_bih1, nullptr,
                                              hc1, hbf1c, wDhh1, dec_bhh1, hn1,
                                              hbf1n);
    fc_argmax_kernel<<<NBLK_FC, 256, 0, stream>>>(hbf1n, wFc, fc_b, out, i + 1,
                                                  pvals, pidxs);
    if (i < Tt - 2) {
      argmax_reduce_kernel<<<1, 32, 0, stream>>>(pvals, pidxs, tok);
      embed_tok_kernel<<<1, 256, 0, stream>>>(tok, dec_emb, xdec);
    }
    // rotate layer-0 bf16 buffers: cur <- just written, next <- spare
    hbf0c = hbf0n; hbf0n = hbf0other; hbf0other = (u16t*)hbf0c;
    { float* t2 = hc;  hc  = hn;  hn  = t2; }
    { float* t2 = hc1; hc1 = hn1; hn1 = t2; }
    { u16t* t3 = hbf1c; hbf1c = hbf1n; hbf1n = t3; }
  }
}
